// HeteroGNN_88991722373486
// MI455X (gfx1250) — matrix operand order repaired and checked by reading, not compiled, not run on hardware
//
#include <hip/hip_runtime.h>
#include <hip/hip_bf16.h>

// ---------------------------------------------------------------------------
// HeteroGNN (2-layer GATv2) for gfx1250 / MI455X.
// Dense projections via v_wmma_f32_16x16x32_f16 (f32 accumulate), edge
// softmax/aggregation via wave32-per-edge kernels with L2 float atomics.
// ---------------------------------------------------------------------------

#define N_A 200000
#define N_B 250000
#define N_P 120000
#define HDIM 128
#define ODIM 64
#define E_AB 400000
#define E_BA 400000
#define E_BP 200000
#define E_PB 200000
#define NEG_SLOPE 0.2f

typedef __attribute__((ext_vector_type(16))) _Float16 v16h;
typedef __attribute__((ext_vector_type(8)))  float    v8f;

// monotonic float<->uint encoding for atomicMax-based segment max
__device__ __forceinline__ unsigned encf(float f) {
  unsigned u = __float_as_uint(f);
  return (u & 0x80000000u) ? ~u : (u | 0x80000000u);
}
__device__ __forceinline__ float decf(unsigned u) {
  unsigned v = (u & 0x80000000u) ? (u & 0x7FFFFFFFu) : ~u;
  return __uint_as_float(v);
}

// ---- flat fp32 -> f16 convert ---------------------------------------------
__global__ void k_f32_to_f16(const float* __restrict__ in,
                             _Float16* __restrict__ out, int n) {
  int i = blockIdx.x * blockDim.x + threadIdx.x;
  if (i < n) out[i] = (_Float16)in[i];
}

// ---- weight prep: W[K,N] fp32 row-major -> Wt[N,Kp] f16 (K zero-padded) ---
__global__ void k_prep_wt(const float* __restrict__ W, _Float16* __restrict__ Wt,
                          int K, int Kp, int N) {
  int i = blockIdx.x * blockDim.x + threadIdx.x;
  if (i >= N * Kp) return;
  int n = i / Kp, k = i - n * Kp;
  Wt[i] = (_Float16)((k < K) ? W[(size_t)k * N + n] : 0.f);
}

// ---- WMMA GEMM: C[M,N] = f32( A_f16[M,K] @ W ) + bias ---------------------
// A row-major f16 (stride K). Wt is [N,Kp] f16 (transposed, padded).
// 8 waves/block: wave w -> columns [16w,16w+16), block -> 16 rows.
__global__ __launch_bounds__(256)
void k_gemm_wmma(const _Float16* __restrict__ A, const _Float16* __restrict__ Wt,
                 const float* __restrict__ bias, float* __restrict__ C,
                 int M, int K, int Kp, int N) {
  __shared__ _Float16 As[16][136];   // Kp <= 128
  const int tid  = threadIdx.x;
  const int row0 = blockIdx.x * 16;

  // stage 16 x Kp tile of A into LDS (zero-padded)
  for (int i = tid; i < 16 * Kp; i += 256) {
    int r = i / Kp, k = i - r * Kp;
    int gr = row0 + r;
    float v = (gr < M && k < K) ? (float)A[(size_t)gr * K + k] : 0.f;
    As[r][k] = (_Float16)v;
  }
  __syncthreads();

  const int wave = tid >> 5;
  const int lane = tid & 31;
  const int colTile = wave;
  if (colTile * 16 >= N) return;          // wave-uniform: EXEC stays all-ones

  const int half = lane >> 4;             // 0: lanes 0-15, 1: lanes 16-31
  const int l16  = lane & 15;

  v8f c = {};
  for (int k0 = 0; k0 < Kp; k0 += 32) {
    v16h a, b;
    // A 16x32 f16 layout: elems 0..7 -> K = k0 + 8*half + e
    //                     elems 8..15 -> K = k0 + 16 + 8*half + (e-8)
#pragma unroll
    for (int e = 0; e < 8; ++e) a[e]     = As[l16][k0 + half * 8 + e];
#pragma unroll
    for (int e = 0; e < 8; ++e) a[8 + e] = As[l16][k0 + 16 + half * 8 + e];
    // B 32x16 f16 layout: col = l16, same K order as A (elems 0..7 -> k0+8*half+e, 8..15 -> k0+16+8*half+(e-8))
    const _Float16* wrow = Wt + (size_t)(colTile * 16 + l16) * Kp + k0 + half * 8;
#pragma unroll
    for (int e = 0; e < 16; ++e) b[e] = wrow[e < 8 ? e : e + 8];

    c = __builtin_amdgcn_wmma_f32_16x16x32_f16(false, a, false, b,
                                               (short)0, c, false, false);
  }

  // C/D layout: vgpr r -> row M = r + 8*half, col = l16
#pragma unroll
  for (int r = 0; r < 8; ++r) {
    int grow = row0 + r + half * 8;
    int gcol = colTile * 16 + l16;
    if (grow < M) {
      float v = c[r];
      if (bias) v += bias[gcol];
      C[(size_t)grow * N + gcol] = v;
    }
  }
}

// ---- edge score: e = leakyrelu(hl[s]+hr[d](+ea@We)) . att ; segment max ---
__global__ __launch_bounds__(256)
void k_edge_score(const float* __restrict__ hl, const float* __restrict__ hr,
                  const int* __restrict__ ei, int E,
                  const float* __restrict__ ea, const float* __restrict__ We,
                  const float* __restrict__ att,
                  float* __restrict__ esc, unsigned* __restrict__ mx) {
  int e = blockIdx.x * 8 + (threadIdx.x >> 5);
  int lane = threadIdx.x & 31;
  if (e >= E) return;                      // wave-uniform
  int s = ei[e], d = ei[E + e];

  const float4 va = *(const float4*)(hl + (size_t)s * HDIM + lane * 4);
  const float4 vb = *(const float4*)(hr + (size_t)d * HDIM + lane * 4);
  float m0 = va.x + vb.x, m1 = va.y + vb.y, m2 = va.z + vb.z, m3 = va.w + vb.w;

  if (We) {
    float ev[8];
#pragma unroll
    for (int k = 0; k < 8; ++k) ev[k] = ea[(size_t)e * 8 + k];
#pragma unroll
    for (int k = 0; k < 8; ++k) {
      const float* wr = We + k * HDIM + lane * 4;
      m0 += ev[k] * wr[0]; m1 += ev[k] * wr[1];
      m2 += ev[k] * wr[2]; m3 += ev[k] * wr[3];
    }
  }
  const float4 at = *(const float4*)(att + lane * 4);
  float v = 0.f;
  v += (m0 > 0.f ? m0 : NEG_SLOPE * m0) * at.x;
  v += (m1 > 0.f ? m1 : NEG_SLOPE * m1) * at.y;
  v += (m2 > 0.f ? m2 : NEG_SLOPE * m2) * at.z;
  v += (m3 > 0.f ? m3 : NEG_SLOPE * m3) * at.w;
#pragma unroll
  for (int off = 16; off > 0; off >>= 1) v += __shfl_xor(v, off, 32);
  if (lane == 0) {
    esc[e] = v;
    atomicMax(&mx[d], encf(v));
  }
}

// ---- exp(e - mx[d]) and segment sum ---------------------------------------
__global__ void k_edge_exp(const int* __restrict__ ei, int E,
                           float* __restrict__ esc,
                           const unsigned* __restrict__ mx,
                           float* __restrict__ den) {
  int e = blockIdx.x * blockDim.x + threadIdx.x;
  if (e >= E) return;
  int d = ei[E + e];
  float ex = __expf(esc[e] - decf(mx[d]));
  esc[e] = ex;
  atomicAdd(&den[d], ex);
}

// ---- scatter: acc[d] += hl[s] * alpha -------------------------------------
__global__ __launch_bounds__(256)
void k_edge_scatter(const float* __restrict__ hl, const int* __restrict__ ei,
                    int E, const float* __restrict__ esc,
                    const float* __restrict__ den, float* __restrict__ acc) {
  int e = blockIdx.x * 8 + (threadIdx.x >> 5);
  int lane = threadIdx.x & 31;
  if (e >= E) return;
  int s = ei[e], d = ei[E + e];
  float alpha = esc[e] / (den[d] + 1e-16f);
  const float4 va = *(const float4*)(hl + (size_t)s * HDIM + lane * 4);
  float* p = acc + (size_t)d * HDIM + lane * 4;
  atomicAdd(p + 0, va.x * alpha);
  atomicAdd(p + 1, va.y * alpha);
  atomicAdd(p + 2, va.z * alpha);
  atomicAdd(p + 3, va.w * alpha);
}

// ---- bias + relu + convert to f16 features --------------------------------
__global__ void k_bias_relu(const float* __restrict__ acc,
                            const float* __restrict__ b1,
                            const float* __restrict__ b2,
                            _Float16* __restrict__ out, int total) {
  int i = blockIdx.x * blockDim.x + threadIdx.x;
  if (i >= total) return;
  int h = i & (HDIM - 1);
  float v = acc[i];
  if (b1) v += b1[h];
  if (b2) v += b2[h];
  out[i] = (_Float16)(v > 0.f ? v : 0.f);
}

// ---------------------------------------------------------------------------
struct RelP { const float *We, *Wl, *Wr, *att, *bias, *bl, *br; };

extern "C" void kernel_launch(void* const* d_in, const int* in_sizes, int n_in,
                              void* d_out, int out_size, void* d_ws, size_t ws_size,
                              hipStream_t stream) {
  const float* x_a  = (const float*)d_in[0];
  const float* x_b  = (const float*)d_in[1];
  const float* x_p  = (const float*)d_in[2];
  const int* ei_ab  = (const int*)d_in[3];
  const int* ei_ba  = (const int*)d_in[4];
  const int* ei_bp  = (const int*)d_in[5];
  const int* ei_pb  = (const int*)d_in[6];
  const float* ea_ab = (const float*)d_in[7];
  const float* ea_ba = (const float*)d_in[8];

  // params: pytree (sorted-dict) flatten order.
  // per relation (ab,ba have We): We?, Wl, Wr, att, bias, bl, br
  // relations sorted: ab, ba, bp, pb ; then lin a/b/p each (W, b)
  int pi = 9;
  RelP conv[2][4];
  for (int l = 0; l < 2; ++l)
    for (int r = 0; r < 4; ++r) {
      RelP p{};
      if (r < 2) p.We = (const float*)d_in[pi++];
      p.Wl   = (const float*)d_in[pi++];
      p.Wr   = (const float*)d_in[pi++];
      p.att  = (const float*)d_in[pi++];
      p.bias = (const float*)d_in[pi++];
      p.bl   = (const float*)d_in[pi++];
      p.br   = (const float*)d_in[pi++];
      conv[l][r] = p;
    }
  const float *linW[3], *linB[3];
  for (int j = 0; j < 3; ++j) {
    linW[j] = (const float*)d_in[pi++];
    linB[j] = (const float*)d_in[pi++];
  }

  // workspace carve (aligned 256B)
  char* ws = (char*)d_ws;
  size_t off = 0;
  auto carve = [&](size_t bytes) -> void* {
    void* p = ws + off;
    off = (off + bytes + 255) & ~(size_t)255;
    return p;
  };
  _Float16* xa16  = (_Float16*)carve((size_t)N_A * 32 * 2);
  _Float16* xb16  = (_Float16*)carve((size_t)N_B * 64 * 2);
  _Float16* xp16  = (_Float16*)carve((size_t)N_P * 16 * 2);
  _Float16* featA = (_Float16*)carve((size_t)N_A * HDIM * 2);
  _Float16* featB = (_Float16*)carve((size_t)N_B * HDIM * 2);
  _Float16* featP = (_Float16*)carve((size_t)N_P * HDIM * 2);
  float* accA = (float*)carve((size_t)N_A * HDIM * 4);
  float* accB = (float*)carve((size_t)N_B * HDIM * 4);
  float* accP = (float*)carve((size_t)N_P * HDIM * 4);
  float* hl   = (float*)carve((size_t)N_B * HDIM * 4);   // max src nodes
  float* hr   = (float*)carve((size_t)N_B * HDIM * 4);   // max dst nodes
  float* esc  = (float*)carve((size_t)E_AB * 4);
  unsigned* mx = (unsigned*)carve((size_t)N_B * 4);
  float* den  = (float*)carve((size_t)N_B * 4);
  _Float16* Wt = (_Float16*)carve((size_t)HDIM * HDIM * 2);

  auto cvt = [&](const float* in, _Float16* out, int n) {
    k_f32_to_f16<<<(n + 255) / 256, 256, 0, stream>>>(in, out, n);
  };
  cvt(x_a, xa16, N_A * 32);
  cvt(x_b, xb16, N_B * 64);
  cvt(x_p, xp16, N_P * 16);

  auto kpad = [](int K) { return (K + 31) & ~31; };

  auto run_gat = [&](const _Float16* xs, int Ks, int Ns,
                     const _Float16* xd, int Kd, int Nd,
                     const int* ei, int E, const float* ea,
                     const RelP& p, float* acc) {
    int Kps = kpad(Ks);
    k_prep_wt<<<(HDIM * Kps + 255) / 256, 256, 0, stream>>>(p.Wl, Wt, Ks, Kps, HDIM);
    k_gemm_wmma<<<(Ns + 15) / 16, 256, 0, stream>>>(xs, Wt, p.bl, hl, Ns, Ks, Kps, HDIM);
    int Kpd = kpad(Kd);
    k_prep_wt<<<(HDIM * Kpd + 255) / 256, 256, 0, stream>>>(p.Wr, Wt, Kd, Kpd, HDIM);
    k_gemm_wmma<<<(Nd + 15) / 16, 256, 0, stream>>>(xd, Wt, p.br, hr, Nd, Kd, Kpd, HDIM);
    hipMemsetAsync(mx, 0, (size_t)Nd * 4, stream);   // enc order: 0 == smallest
    hipMemsetAsync(den, 0, (size_t)Nd * 4, stream);
    k_edge_score<<<(E + 7) / 8, 256, 0, stream>>>(hl, hr, ei, E, ea,
                                                  ea ? p.We : nullptr, p.att, esc, mx);
    k_edge_exp<<<(E + 255) / 256, 256, 0, stream>>>(ei, E, esc, mx, den);
    k_edge_scatter<<<(E + 7) / 8, 256, 0, stream>>>(hl, ei, E, esc, den, acc);
  };

  const _Float16 *ia = xa16, *ib = xb16, *ip = xp16;
  int Ka = 32, Kb = 64, Kp_ = 16;
  for (int l = 0; l < 2; ++l) {
    hipMemsetAsync(accA, 0, (size_t)N_A * HDIM * 4, stream);
    hipMemsetAsync(accB, 0, (size_t)N_B * HDIM * 4, stream);
    hipMemsetAsync(accP, 0, (size_t)N_P * HDIM * 4, stream);

    run_gat(ia, Ka,  N_A, ib, Kb,  N_B, ei_ab, E_AB, ea_ab,  conv[l][0], accB); // ab
    run_gat(ip, Kp_, N_P, ib, Kb,  N_B, ei_pb, E_PB, nullptr, conv[l][3], accB); // pb
    run_gat(ib, Kb,  N_B, ia, Ka,  N_A, ei_ba, E_BA, ea_ba,  conv[l][1], accA); // ba
    run_gat(ib, Kb,  N_B, ip, Kp_, N_P, ei_bp, E_BP, nullptr, conv[l][2], accP); // bp

    k_bias_relu<<<((N_A * HDIM) + 255) / 256, 256, 0, stream>>>(
        accA, conv[l][1].bias, nullptr, featA, N_A * HDIM);
    k_bias_relu<<<((N_B * HDIM) + 255) / 256, 256, 0, stream>>>(
        accB, conv[l][0].bias, conv[l][3].bias, featB, N_B * HDIM);
    k_bias_relu<<<((N_P * HDIM) + 255) / 256, 256, 0, stream>>>(
        accP, conv[l][2].bias, nullptr, featP, N_P * HDIM);

    ia = featA; ib = featB; ip = featP;
    Ka = Kb = Kp_ = HDIM;
  }

  // final per-type linears -> d_out (fp32, concatenated a,b,p)
  float* outA = (float*)d_out;
  float* outB = outA + (size_t)N_A * ODIM;
  float* outP = outB + (size_t)N_B * ODIM;

  k_prep_wt<<<(ODIM * HDIM + 255) / 256, 256, 0, stream>>>(linW[0], Wt, HDIM, HDIM, ODIM);
  k_gemm_wmma<<<(N_A + 15) / 16, 256, 0, stream>>>(featA, Wt, linB[0], outA, N_A, HDIM, HDIM, ODIM);
  k_prep_wt<<<(ODIM * HDIM + 255) / 256, 256, 0, stream>>>(linW[1], Wt, HDIM, HDIM, ODIM);
  k_gemm_wmma<<<(N_B + 15) / 16, 256, 0, stream>>>(featB, Wt, linB[1], outB, N_B, HDIM, HDIM, ODIM);
  k_prep_wt<<<(ODIM * HDIM + 255) / 256, 256, 0, stream>>>(linW[2], Wt, HDIM, HDIM, ODIM);
  k_gemm_wmma<<<(N_P + 15) / 16, 256, 0, stream>>>(featP, Wt, linB[2], outP, N_P, HDIM, HDIM, ODIM);
}
